// DeepseekV2DecoderLayer_29600914604513
// MI455X (gfx1250) — matrix operand order repaired and checked by reading, not compiled, not run on hardware
//
#include <hip/hip_runtime.h>

// ---------------------------------------------------------------------------
// DeepSeek-V2-style decoder layer for gfx1250 (MI455X), wave32 + WMMA bf16.
// ---------------------------------------------------------------------------

#define T_TOK 1024
#define H_DIM 2048
#define NH    16
#define DN    128
#define DR    64
#define DQ    192      // DN + DR
#define DV    128
#define QL    1536
#define KLAT  512
#define INH   16
#define IHD   128
#define TOPK  256
#define NE    8
#define MI_D  1024
#define SI_D  1024
#define QKV_N (QL + KLAT + DR)   // 2112
#define EPSV  1e-6f
#define ATT_SCALE 0.07216878364870323f   // 192^-0.5
#define IDX_SCALE 0.08838834764831845f   // 128^-0.5
#define INH_SCALE 0.25f                  // 16^-0.5
#define FP8_MAX 448.0f
#define INF_F (__builtin_inff())

typedef __attribute__((ext_vector_type(16))) __bf16 v16bf;
typedef __attribute__((ext_vector_type(8)))  __bf16 v8bf;
typedef __attribute__((ext_vector_type(4)))  __bf16 v4bf;
typedef __attribute__((ext_vector_type(8)))  float  v8f;

#define WMMA_BF16(a, b, c) \
  __builtin_amdgcn_wmma_f32_16x16x32_bf16(false, (a), false, (b), (short)0, (c), false, false)

__device__ __forceinline__ v4bf pack4(float a, float b, float c, float d) {
  v4bf v;
  v[0] = (__bf16)a; v[1] = (__bf16)b; v[2] = (__bf16)c; v[3] = (__bf16)d;
  return v;
}

// ---- WMMA fragment gathers: contiguous 16B LDS reads + shufflevector ------
// A (16x32 bf16) from row-major [M][K] LDS (stride = multiple of 8 halves):
//   lanes 0-15 row=lane kOff=0, lanes 16-31 same rows kOff=8;
//   halves {kOff..kOff+7} and {16+kOff..23+kOff}.
__device__ __forceinline__ v16bf fragA(const __bf16* base, int stride,
                                       int rowBase, int kBase, int lane) {
  const __bf16* p = base + (rowBase + (lane & 15)) * stride + kBase +
                    ((lane < 16) ? 0 : 8);
  v8bf lo = *(const v8bf*)p;
  v8bf hi = *(const v8bf*)(p + 16);
  return __builtin_shufflevector(lo, hi, 0, 1, 2, 3, 4, 5, 6, 7,
                                 8, 9, 10, 11, 12, 13, 14, 15);
}
// B (32x16 bf16) from TRANSPOSED LDS [N][K]: lane col = lane&15,
//   halves {kOff..kOff+7} and {16+kOff..23+kOff}, kOff = 0 (lanes 0-15) or 8 (lanes 16-31).
__device__ __forceinline__ v16bf fragBT(const __bf16* base, int stride,
                                        int kBase, int colBase, int lane) {
  const __bf16* p = base + (colBase + (lane & 15)) * stride + kBase +
                    ((lane < 16) ? 0 : 8);
  v8bf lo = *(const v8bf*)p;
  v8bf hi = *(const v8bf*)(p + 16);
  return __builtin_shufflevector(lo, hi, 0, 1, 2, 3, 4, 5, 6, 7,
                                 8, 9, 10, 11, 12, 13, 14, 15);
}

// ---------------------------------------------------------------------------
// Generic tiled GEMM: C[M,N] = A[M,K] @ B[K,N] (+residual/+C, *rowScale).
// f32 in/out, bf16 WMMA compute; 128x128 block, 8 waves, K-step 32.
// Interior blocks: float4 global loads, register-staged double buffering.
// ---------------------------------------------------------------------------
__device__ __forceinline__ void gemm_compute(const __bf16* sA, const __bf16* sBt,
                                             v8f (&acc)[2][4], int wr, int wc,
                                             int lane) {
  v16bf afr[2], bfr[4];
#pragma unroll
  for (int mt = 0; mt < 2; ++mt) afr[mt] = fragA(sA, 40, wr + mt * 16, 0, lane);
#pragma unroll
  for (int nt = 0; nt < 4; ++nt) bfr[nt] = fragBT(sBt, 40, 0, wc + nt * 16, lane);
#pragma unroll
  for (int mt = 0; mt < 2; ++mt)
#pragma unroll
    for (int nt = 0; nt < 4; ++nt)
      acc[mt][nt] = WMMA_BF16(afr[mt], bfr[nt], acc[mt][nt]);
}

__global__ __launch_bounds__(256) void gemm_kernel(
    const float* __restrict__ A, int lda,
    const float* __restrict__ B, int ldb,
    float* __restrict__ C, int ldc,
    int M, int N, int K,
    const float* __restrict__ residual,
    const float* __restrict__ rowScale, int rowScaleStride,
    int accumulate) {
  __shared__ __align__(16) __bf16 sA[128][40];    // [M][K]
  __shared__ __align__(16) __bf16 sBt[128][40];   // [N][K] (transposed)
  const int tid = threadIdx.x;
  const int m0 = blockIdx.y * 128;
  const int n0 = blockIdx.x * 128;
  const int wave = tid >> 5, lane = tid & 31;
  const int wr = (wave >> 1) * 32;
  const int wc = (wave & 1) * 64;
  const int cHi = (lane < 16) ? 0 : 8;
  const int cN = lane & 15;

  v8f acc[2][4];
#pragma unroll
  for (int mt = 0; mt < 2; ++mt)
#pragma unroll
    for (int nt = 0; nt < 4; ++nt)
#pragma unroll
      for (int r = 0; r < 8; ++r) acc[mt][nt][r] = 0.0f;

  const bool fast = (m0 + 128 <= M) && (n0 + 128 <= N) && ((K & 31) == 0);
  if (fast) {
    // Register-staged pipeline: load tile k+1 from global while computing k.
    float4 aReg[4];
    float bReg[4][4];
    auto loadA = [&](int k0) {
#pragma unroll
      for (int i = 0; i < 4; ++i) {
        int e = i * 256 + tid;            // 128 rows x 8 k-groups
        int r = e >> 3, kg = e & 7;
        aReg[i] = *(const float4*)&A[(long long)(m0 + r) * lda + k0 + kg * 4];
      }
    };
    auto loadB = [&](int k0) {
#pragma unroll
      for (int i = 0; i < 4; ++i) {
        int e = i * 256 + tid;            // 128 cols x 8 k-groups
        int c = e & 127, kg = e >> 7;
#pragma unroll
        for (int j = 0; j < 4; ++j)
          bReg[i][j] = B[(long long)(k0 + kg * 4 + j) * ldb + n0 + c];
      }
    };
    auto storeStage = [&]() {
#pragma unroll
      for (int i = 0; i < 4; ++i) {
        int e = i * 256 + tid;
        int r = e >> 3, kg = e & 7;
        *(v4bf*)&sA[r][kg * 4] = pack4(aReg[i].x, aReg[i].y, aReg[i].z, aReg[i].w);
      }
#pragma unroll
      for (int i = 0; i < 4; ++i) {
        int e = i * 256 + tid;
        int c = e & 127, kg = e >> 7;
        *(v4bf*)&sBt[c][kg * 4] = pack4(bReg[i][0], bReg[i][1], bReg[i][2], bReg[i][3]);
      }
    };
    loadA(0);
    loadB(0);
    for (int k0 = 0; k0 < K; k0 += 32) {
      storeStage();
      __syncthreads();
      if (k0 + 32 < K) { loadA(k0 + 32); loadB(k0 + 32); }
      gemm_compute(&sA[0][0], &sBt[0][0], acc, wr, wc, lane);
      __syncthreads();
    }
  } else {
    for (int k0 = 0; k0 < K; k0 += 32) {
#pragma unroll
      for (int i = 0; i < 16; ++i) {      // A tile 128x32
        int e = i * 256 + tid;
        int r = e >> 5, kk = e & 31;
        int gr = m0 + r, gk = k0 + kk;
        float v = (gr < M && gk < K) ? A[(long long)gr * lda + gk] : 0.0f;
        sA[r][kk] = (__bf16)v;
      }
#pragma unroll
      for (int i = 0; i < 16; ++i) {      // B tile 32x128 -> transposed store
        int e = i * 256 + tid;
        int kk = e >> 7, c = e & 127;
        int gk = k0 + kk, gc = n0 + c;
        float v = (gk < K && gc < N) ? B[(long long)gk * ldb + gc] : 0.0f;
        sBt[c][kk] = (__bf16)v;
      }
      __syncthreads();
      gemm_compute(&sA[0][0], &sBt[0][0], acc, wr, wc, lane);
      __syncthreads();
    }
  }

#pragma unroll
  for (int mt = 0; mt < 2; ++mt)
#pragma unroll
    for (int nt = 0; nt < 4; ++nt)
#pragma unroll
      for (int r = 0; r < 8; ++r) {
        int row = m0 + wr + mt * 16 + r + cHi;
        int col = n0 + wc + nt * 16 + cN;
        if (row < M && col < N) {
          float v = acc[mt][nt][r];
          if (rowScale) v *= rowScale[(long long)row * rowScaleStride];
          if (residual) v += residual[(long long)row * ldc + col];
          if (accumulate) v += C[(long long)row * ldc + col];
          C[(long long)row * ldc + col] = v;
        }
      }
}

// ---------------------------------------------------------------------------
// Elementwise / normalization kernels
// ---------------------------------------------------------------------------
__global__ __launch_bounds__(256) void rmsnorm_kernel(
    const float* __restrict__ in, int inStride, const float* __restrict__ w,
    float* __restrict__ out, int outStride, int dim) {
  const int row = blockIdx.x;
  const float* x = in + (long long)row * inStride;
  __shared__ float red[256];
  float s = 0.0f;
  for (int i = threadIdx.x; i < dim; i += 256) { float v = x[i]; s += v * v; }
  red[threadIdx.x] = s;
  __syncthreads();
  for (int off = 128; off > 0; off >>= 1) {
    if (threadIdx.x < off) red[threadIdx.x] += red[threadIdx.x + off];
    __syncthreads();
  }
  float r = rsqrtf(red[0] / dim + EPSV);
  for (int i = threadIdx.x; i < dim; i += 256)
    out[(long long)row * outStride + i] = x[i] * w[i] * r;
}

__global__ __launch_bounds__(128) void layernorm128_kernel(
    const float* __restrict__ in, const float* __restrict__ w,
    const float* __restrict__ b, float* __restrict__ out) {
  const int row = blockIdx.x;
  const int i = threadIdx.x;  // dim == 128
  __shared__ float red[128];
  float v = in[(long long)row * 128 + i];
  red[i] = v; __syncthreads();
  for (int off = 64; off > 0; off >>= 1) {
    if (i < off) red[i] += red[i + off];
    __syncthreads();
  }
  float mean = red[0] / 128.0f;
  __syncthreads();
  float d = v - mean;
  red[i] = d * d; __syncthreads();
  for (int off = 64; off > 0; off >>= 1) {
    if (i < off) red[i] += red[i + off];
    __syncthreads();
  }
  float var = red[0] / 128.0f;
  out[(long long)row * 128 + i] = d * rsqrtf(var + EPSV) * w[i] + b[i];
}

__global__ void rope_kernel(float* __restrict__ x, const int* __restrict__ pos,
                            int heads, int rowStride, int headStride,
                            int offset, int D, int total) {
  int idx = blockIdx.x * blockDim.x + threadIdx.x;
  if (idx >= total) return;
  int half = D >> 1;
  int i = idx % half;
  int th = idx / half;
  int h = th % heads;
  int t = th / heads;
  float* p = x + (long long)t * rowStride + h * headStride + offset + 2 * i;
  float ang = (float)pos[t] * powf(10000.0f, -(2.0f * i) / (float)D);
  float c = cosf(ang), s = sinf(ang);
  float x1 = p[0], x2 = p[1];
  p[0] = x1 * c - x2 * s;
  p[1] = x1 * s + x2 * c;
}

// wts[t,h] = (hp[t,h]+b[h]) * exp2(ceil(log2(max(amax,1e-12)/448))) * IDX_SCALE/4
__global__ __launch_bounds__(256) void wts_kernel(
    const float* __restrict__ iq, const float* __restrict__ wp_raw,
    const float* __restrict__ wp_b, float* __restrict__ wts) {
  const int t = blockIdx.x;
  const int h = threadIdx.x >> 4, j = threadIdx.x & 15;
  const float* p = iq + ((long long)t * INH + h) * IHD;
  float m = 0.0f;
  for (int i = j; i < IHD; i += 16) m = fmaxf(m, fabsf(p[i]));
  __shared__ float red[16][16];
  red[h][j] = m; __syncthreads();
  if (j == 0) {
    float am = 0.0f;
#pragma unroll
    for (int i = 0; i < 16; ++i) am = fmaxf(am, red[h][i]);
    float qs = exp2f(ceilf(log2f(fmaxf(am, 1e-12f) / FP8_MAX)));
    wts[t * INH + h] = (wp_raw[t * INH + h] + wp_b[h]) * qs * IDX_SCALE * INH_SCALE;
  }
}

// ---------------------------------------------------------------------------
// Indexer scores: s[t,j] = sum_h wts[t,h] * relu(iq[t,h,:] . ik[j,:]); causal.
// ---------------------------------------------------------------------------
__global__ __launch_bounds__(256) void idx_score_kernel(
    const float* __restrict__ iq, const float* __restrict__ ik,
    const float* __restrict__ wts, float* __restrict__ s) {
  const int t0 = blockIdx.x * 32;
  const int j0 = blockIdx.y * 128;
  const int tid = threadIdx.x, wave = tid >> 5, lane = tid & 31;
  const int cN = lane & 15, cHi = (lane < 16) ? 0 : 8;

  if (j0 > t0 + 31) {
    for (int i = tid; i < 32 * 128; i += 256) {
      int r = i >> 7, c = i & 127;
      s[(long long)(t0 + r) * T_TOK + j0 + c] = -INF_F;
    }
    return;
  }

  __shared__ __align__(16) __bf16 sIK[128][136];   // [j][d] -> B^T source
  __shared__ __align__(16) __bf16 sIQ[32][136];    // [t][d] -> A source
  __shared__ float sW[32][16];

#pragma unroll
  for (int it = 0; it < 16; ++it) {     // 128x128 f32, float4 granules
    int e = it * 256 + tid;
    int c4 = e & 31, r = e >> 5;
    float4 v = *(const float4*)&ik[(long long)(j0 + r) * IHD + c4 * 4];
    *(v4bf*)&sIK[r][c4 * 4] = pack4(v.x, v.y, v.z, v.w);
  }
  for (int i = tid; i < 32 * 16; i += 256)
    sW[i >> 4][i & 15] = wts[(t0 + (i >> 4)) * INH + (i & 15)];

  float accS[2][8];
#pragma unroll
  for (int mt = 0; mt < 2; ++mt)
#pragma unroll
    for (int r = 0; r < 8; ++r) accS[mt][r] = 0.0f;

  for (int h = 0; h < INH; ++h) {
#pragma unroll
    for (int it = 0; it < 4; ++it) {    // 32x128 f32
      int e = it * 256 + tid;
      int c4 = e & 31, r = e >> 5;
      float4 v = *(const float4*)&iq[(((long long)(t0 + r)) * INH + h) * IHD + c4 * 4];
      *(v4bf*)&sIQ[r][c4 * 4] = pack4(v.x, v.y, v.z, v.w);
    }
    __syncthreads();
    v8f cc[2];
#pragma unroll
    for (int mt = 0; mt < 2; ++mt)
#pragma unroll
      for (int r = 0; r < 8; ++r) cc[mt][r] = 0.0f;
#pragma unroll
    for (int kb = 0; kb < IHD; kb += 32) {
      v16bf bfr = fragBT(&sIK[0][0], 136, kb, wave * 16, lane);
#pragma unroll
      for (int mt = 0; mt < 2; ++mt) {
        v16bf afr = fragA(&sIQ[0][0], 136, mt * 16, kb, lane);
        cc[mt] = WMMA_BF16(afr, bfr, cc[mt]);
      }
    }
#pragma unroll
    for (int mt = 0; mt < 2; ++mt)
#pragma unroll
      for (int r = 0; r < 8; ++r)
        accS[mt][r] += sW[mt * 16 + r + cHi][h] * fmaxf(cc[mt][r], 0.0f);
    __syncthreads();
  }

#pragma unroll
  for (int mt = 0; mt < 2; ++mt)
#pragma unroll
    for (int r = 0; r < 8; ++r) {
      int row = t0 + mt * 16 + r + cHi;
      int col = j0 + wave * 16 + cN;
      s[(long long)row * T_TOK + col] = (col <= row) ? accS[mt][r] : -INF_F;
    }
}

// ---------------------------------------------------------------------------
// Top-k(256) per row -> byte mask.  Radix threshold search on monotone keys.
// ---------------------------------------------------------------------------
__device__ __forceinline__ unsigned f2u_mono(float f) {
  unsigned u = __float_as_uint(f);
  return (u & 0x80000000u) ? ~u : (u | 0x80000000u);
}

__global__ __launch_bounds__(256) void topk_kernel(
    const float* __restrict__ s, unsigned char* __restrict__ allowed) {
  const int t = blockIdx.x;
  const int n = t + 1;
  unsigned char* arow = allowed + (long long)t * T_TOK;
  const float* srow = s + (long long)t * T_TOK;
  for (int j = threadIdx.x; j < T_TOK; j += 256) arow[j] = 0;
  if (n <= TOPK) {
    for (int j = threadIdx.x; j < n; j += 256) arow[j] = 1;
    return;
  }
  __shared__ unsigned su[T_TOK];
  __shared__ int cnt[256];
  for (int j = threadIdx.x; j < n; j += 256) su[j] = f2u_mono(srow[j]);
  __syncthreads();
  unsigned thr = 0;
  for (int bit = 31; bit >= 0; --bit) {
    unsigned cand = thr | (1u << bit);
    int c = 0;
    for (int j = threadIdx.x; j < n; j += 256) c += (su[j] >= cand) ? 1 : 0;
    cnt[threadIdx.x] = c; __syncthreads();
    for (int off = 128; off > 0; off >>= 1) {
      if (threadIdx.x < off) cnt[threadIdx.x] += cnt[threadIdx.x + off];
      __syncthreads();
    }
    if (cnt[0] >= TOPK) thr = cand;
    __syncthreads();
  }
  for (int j = threadIdx.x; j < n; j += 256) arow[j] = (su[j] >= thr) ? 1 : 0;
}

// ---------------------------------------------------------------------------
// Flash attention: one workgroup per (head, 32-query block), 64-key tiles.
// ---------------------------------------------------------------------------
__global__ __launch_bounds__(256) void attn_kernel(
    const float* __restrict__ q,        // [T][NH*DQ], pe already roped
    const float* __restrict__ kv,       // [T][NH*(DN+DV)]
    const float* __restrict__ qkv,      // k_pe at col 2048, stride 2112, roped
    const unsigned char* __restrict__ allowed,
    float* __restrict__ o) {            // [T][NH*DV]
  const int h = blockIdx.x;
  const int q0 = blockIdx.y * 32;
  const int tid = threadIdx.x, wave = tid >> 5, lane = tid & 31;
  const int cN = lane & 15, cHi = (lane < 16) ? 0 : 8;
  const int mtS = wave >> 2, ntS = wave & 3;

  __shared__ __align__(16) __bf16 sK[64][200];    // [j][d] -> B^T for QK^T
  __shared__ __align__(16) __bf16 sVt[128][72];   // [d][j] -> B^T for PV
  __shared__ __align__(16) __bf16 sP[32][72];     // probs bf16 -> A for PV
  __shared__ float  sS[32][72];                   // raw scores
  __shared__ float  sM[32], sL[32], sAl[32];

  // Q fragments for this wave's S row-tile (rows q0+mtS*16..+15), K = 192.
  v16bf qf[6];
  {
    int r = lane & 15;
    int kOff = (lane < 16) ? 0 : 8;
    int row = q0 + mtS * 16 + r;
    const float* qp = q + (long long)row * (NH * DQ) + h * DQ;
#pragma unroll
    for (int kb = 0; kb < 6; ++kb) {
      const float* p0 = qp + kb * 32 + kOff;
#pragma unroll
      for (int i = 0; i < 8; ++i) {
        qf[kb][i]     = (__bf16)(p0[i] * ATT_SCALE);
        qf[kb][i + 8] = (__bf16)(p0[16 + i] * ATT_SCALE);
      }
    }
  }

  v8f accO[2];
#pragma unroll
  for (int mt = 0; mt < 2; ++mt)
#pragma unroll
    for (int r = 0; r < 8; ++r) accO[mt][r] = 0.0f;
  if (tid < 32) { sM[tid] = -INF_F; sL[tid] = 0.0f; }
  __syncthreads();

  const int jmax = q0 + 31;
  for (int j0 = 0; j0 <= jmax; j0 += 64) {
    // Stage K tile [j][d]: d<128 k_nope from kv, d>=128 k_pe from qkv.
#pragma unroll
    for (int it = 0; it < 12; ++it) {   // 64 x 48 float4-granules
      int e = it * 256 + tid;
      int d4 = e % 48, jj = e / 48;
      int row = j0 + jj;
      int d = d4 * 4;
      float4 v;
      if (d < DN)
        v = *(const float4*)&kv[(long long)row * (NH * 256) + h * 256 + d];
      else
        v = *(const float4*)&qkv[(long long)row * QKV_N + (QL + KLAT) + (d - DN)];
      *(v4bf*)&sK[jj][d] = pack4(v.x, v.y, v.z, v.w);
    }
    // Stage V transposed [d][j].
#pragma unroll
    for (int it = 0; it < 8; ++it) {    // 128 d x 16 j-groups
      int e = it * 256 + tid;
      int d = e & 127, jg = e >> 7;
      const float* vp = &kv[(long long)(j0 + jg * 4) * (NH * 256) + h * 256 + DN + d];
      float a0 = vp[0];
      float a1 = vp[NH * 256];
      float a2 = vp[2 * NH * 256];
      float a3 = vp[3 * NH * 256];
      *(v4bf*)&sVt[d][jg * 4] = pack4(a0, a1, a2, a3);
    }
    __syncthreads();

    // S tile (16x16 per wave), K = 192.
    v8f sc;
#pragma unroll
    for (int r = 0; r < 8; ++r) sc[r] = 0.0f;
#pragma unroll
    for (int kb = 0; kb < 6; ++kb) {
      v16bf bfr = fragBT(&sK[0][0], 200, kb * 32, ntS * 16, lane);
      sc = WMMA_BF16(qf[kb], bfr, sc);
    }
#pragma unroll
    for (int r = 0; r < 8; ++r) {
      int row = q0 + mtS * 16 + r + cHi;
      int col = j0 + ntS * 16 + cN;
      float v = sc[r];
      if (col > row || !allowed[(long long)row * T_TOK + col]) v = -INF_F;
      sS[mtS * 16 + r + cHi][ntS * 16 + cN] = v;
    }
    __syncthreads();

    // Streaming softmax (one thread per query row); write P as bf16.
    if (tid < 32) {
      float m = sM[tid], l = sL[tid];
      float mrow = m;
      for (int c = 0; c < 64; ++c) mrow = fmaxf(mrow, sS[tid][c]);
      float alpha;
      float psum = 0.0f;
      if (mrow == -INF_F) {
        alpha = 1.0f;
        for (int c = 0; c < 64; ++c) sP[tid][c] = (__bf16)0.0f;
      } else {
        alpha = (m == -INF_F) ? 0.0f : expf(m - mrow);
        for (int c = 0; c < 64; ++c) {
          float sv = sS[tid][c];
          float p = (sv == -INF_F) ? 0.0f : expf(sv - mrow);
          sP[tid][c] = (__bf16)p;
          psum += p;
        }
      }
      sAl[tid] = alpha;
      sM[tid] = mrow;
      sL[tid] = l * alpha + psum;
    }
    __syncthreads();

    // Rescale O accumulators, add P @ V (wave owns V cols [16w,16w+16)).
#pragma unroll
    for (int mt = 0; mt < 2; ++mt)
#pragma unroll
      for (int r = 0; r < 8; ++r) accO[mt][r] *= sAl[mt * 16 + r + cHi];
#pragma unroll
    for (int kb = 0; kb < 2; ++kb) {
      v16bf bfr = fragBT(&sVt[0][0], 72, kb * 32, wave * 16, lane);
#pragma unroll
      for (int mt = 0; mt < 2; ++mt) {
        v16bf afr = fragA(&sP[0][0], 72, mt * 16, kb * 32, lane);
        accO[mt] = WMMA_BF16(afr, bfr, accO[mt]);
      }
    }
    __syncthreads();
  }

#pragma unroll
  for (int mt = 0; mt < 2; ++mt)
#pragma unroll
    for (int r = 0; r < 8; ++r) {
      int row = q0 + mt * 16 + r + cHi;
      float l = sL[mt * 16 + r + cHi];
      float v = (l > 0.0f) ? accO[mt][r] / l : 0.0f;
      o[(long long)row * (NH * DV) + h * DV + wave * 16 + cN] = v;
    }
}

// ---------------------------------------------------------------------------
// MoE gating: softmax over 8 experts, top-2, renormalize -> rw[T][NE].
// ---------------------------------------------------------------------------
__global__ __launch_bounds__(256) void gate_kernel(
    const float* __restrict__ h2, const float* __restrict__ Wg,
    float* __restrict__ rw) {
  const int t = blockIdx.x;
  const int tid = threadIdx.x;
  __shared__ float red[NE][256];
  const float* x = h2 + (long long)t * H_DIM;
  for (int e = 0; e < NE; ++e) {
    float s = 0.0f;
    for (int i = tid; i < H_DIM; i += 256) s += x[i] * Wg[(long long)i * NE + e];
    red[e][tid] = s;
  }
  __syncthreads();
  if (tid < NE) {
    float s = 0.0f;
    for (int i = 0; i < 256; ++i) s += red[tid][i];
    red[tid][0] = s;
  }
  __syncthreads();
  if (tid == 0) {
    float lg[NE], p[NE];
    float m = -INF_F;
    for (int e = 0; e < NE; ++e) { lg[e] = red[e][0]; m = fmaxf(m, lg[e]); }
    float se = 0.0f;
    for (int e = 0; e < NE; ++e) { p[e] = expf(lg[e] - m); se += p[e]; }
    for (int e = 0; e < NE; ++e) p[e] /= se;
    int i1 = 0;
    for (int e = 1; e < NE; ++e) if (p[e] > p[i1]) i1 = e;
    int i2 = (i1 == 0) ? 1 : 0;
    for (int e = 0; e < NE; ++e) if (e != i1 && p[e] > p[i2]) i2 = e;
    float s2 = p[i1] + p[i2];
    for (int e = 0; e < NE; ++e) rw[t * NE + e] = 0.0f;
    rw[t * NE + i1] = p[i1] / s2;
    rw[t * NE + i2] = p[i2] / s2;
  }
}

__global__ void silu_mul_kernel(const float* __restrict__ g,
                                const float* __restrict__ u,
                                float* __restrict__ y, int n) {
  int i = blockIdx.x * blockDim.x + threadIdx.x;
  if (i < n) {
    float gv = g[i];
    y[i] = gv / (1.0f + expf(-gv)) * u[i];
  }
}

__global__ void zero_kernel(float* __restrict__ p, int n) {
  int i = blockIdx.x * blockDim.x + threadIdx.x;
  if (i < n) p[i] = 0.0f;
}

__global__ void add2_kernel(const float* __restrict__ a,
                            const float* __restrict__ b,
                            float* __restrict__ out, int n) {
  int i = blockIdx.x * blockDim.x + threadIdx.x;
  if (i < n) out[i] = a[i] + b[i];
}

// ---------------------------------------------------------------------------
// Host orchestration
// ---------------------------------------------------------------------------
static void gemm(hipStream_t s, const float* A, int lda, const float* B, int ldb,
                 float* C, int ldc, int M, int N, int K,
                 const float* residual = nullptr,
                 const float* rowScale = nullptr, int rss = 0, int accumulate = 0) {
  dim3 g((N + 127) / 128, (M + 127) / 128);
  gemm_kernel<<<g, 256, 0, s>>>(A, lda, B, ldb, C, ldc, M, N, K,
                                residual, rowScale, rss, accumulate);
}

extern "C" void kernel_launch(void* const* d_in, const int* in_sizes, int n_in,
                              void* d_out, int out_size, void* d_ws, size_t ws_size,
                              hipStream_t stream) {
  (void)in_sizes; (void)n_in; (void)out_size; (void)ws_size;
  const int*   positions  = (const int*)d_in[0];
  const float* X          = (const float*)d_in[1];
  const float* input_ln_w = (const float*)d_in[2];
  const float* post_ln_w  = (const float*)d_in[3];
  const float* W_qkv_a    = (const float*)d_in[4];
  const float* q_a_ln_w   = (const float*)d_in[5];
  const float* W_q_b      = (const float*)d_in[6];
  const float* kv_a_ln_w  = (const float*)d_in[7];
  const float* W_kv_b     = (const float*)d_in[8];
  const float* W_o        = (const float*)d_in[9];
  const float* idx_wq_b   = (const float*)d_in[10];
  const float* idx_wk     = (const float*)d_in[11];
  const float* idx_kn_w   = (const float*)d_in[12];
  const float* idx_kn_b   = (const float*)d_in[13];
  const float* idx_wp_w   = (const float*)d_in[14];
  const float* idx_wp_b   = (const float*)d_in[15];
  const float* W_gate     = (const float*)d_in[16];
  const float* We_gate    = (const float*)d_in[17];
  const float* We_up      = (const float*)d_in[18];
  const float* We_down    = (const float*)d_in[19];
  const float* Ws_gate    = (const float*)d_in[20];
  const float* Ws_up      = (const float*)d_in[21];
  const float* Ws_down    = (const float*)d_in[22];

  char* base = (char*)d_ws;
  size_t off = 0;
  auto alloc = [&](size_t bytes) -> void* {
    void* p = base + off;
    off += (bytes + 255) & ~(size_t)255;
    return p;
  };
  const int T = T_TOK;
  float* h       = (float*)alloc((size_t)T * H_DIM * 4);
  float* qkv     = (float*)alloc((size_t)T * QKV_N * 4);
  float* q_cn    = (float*)alloc((size_t)T * QL * 4);
  float* kv_cn   = (float*)alloc((size_t)T * KLAT * 4);
  float* iq      = (float*)alloc((size_t)T * INH * IHD * 4);
  float* ik      = (float*)alloc((size_t)T * IHD * 4);
  float* wp_raw  = (float*)alloc((size_t)T * INH * 4);
  float* wts     = (float*)alloc((size_t)T * INH * 4);
  float* sbuf    = (float*)alloc((size_t)T * T * 4);
  unsigned char* allowed = (unsigned char*)alloc((size_t)T * T);
  float* qb      = (float*)alloc((size_t)T * NH * DQ * 4);
  float* kvb     = (float*)alloc((size_t)T * NH * 256 * 4);
  float* obuf    = (float*)alloc((size_t)T * NH * DV * 4);
  float* x1      = (float*)alloc((size_t)T * H_DIM * 4);
  float* h2      = (float*)alloc((size_t)T * H_DIM * 4);
  float* rw      = (float*)alloc((size_t)T * NE * 4);
  float* gbuf    = (float*)alloc((size_t)T * MI_D * 4);
  float* ubuf    = (float*)alloc((size_t)T * MI_D * 4);
  float* tbuf    = (float*)alloc((size_t)T * MI_D * 4);
  float* Racc    = (float*)alloc((size_t)T * H_DIM * 4);

  // 1. h = rmsnorm(x, input_ln_w)
  rmsnorm_kernel<<<T, 256, 0, stream>>>(X, H_DIM, input_ln_w, h, H_DIM, H_DIM);
  // 2. qkv = h @ W_qkv_a
  gemm(stream, h, H_DIM, W_qkv_a, QKV_N, qkv, QKV_N, T, QKV_N, H_DIM);
  // 3. rope(k_pe) in place (cols [2048,2112))
  {
    int total = T * (DR / 2);
    rope_kernel<<<(total + 255) / 256, 256, 0, stream>>>(
        qkv, positions, 1, QKV_N, 0, QL + KLAT, DR, total);
  }
  // 4/5. q_c, kv_c rmsnorm from qkv slices
  rmsnorm_kernel<<<T, 256, 0, stream>>>(qkv, QKV_N, q_a_ln_w, q_cn, QL, QL);
  rmsnorm_kernel<<<T, 256, 0, stream>>>(qkv + QL, QKV_N, kv_a_ln_w, kv_cn, KLAT, KLAT);
  // 6. iq = q_cn @ idx_wq_b ; 7. rope(iq_pe)
  gemm(stream, q_cn, QL, idx_wq_b, INH * IHD, iq, INH * IHD, T, INH * IHD, QL);
  {
    int total = T * INH * (DR / 2);
    rope_kernel<<<(total + 255) / 256, 256, 0, stream>>>(
        iq, positions, INH, INH * IHD, IHD, 0, DR, total);
  }
  // 8/9/10. ik = rope(layernorm(h @ idx_wk))
  gemm(stream, h, H_DIM, idx_wk, IHD, ik, IHD, T, IHD, H_DIM);
  layernorm128_kernel<<<T, 128, 0, stream>>>(ik, idx_kn_w, idx_kn_b, ik);
  {
    int total = T * (DR / 2);
    rope_kernel<<<(total + 255) / 256, 256, 0, stream>>>(
        ik, positions, 1, IHD, 0, 0, DR, total);
  }
  // 11/12. indexer weights
  gemm(stream, h, H_DIM, idx_wp_w, INH, wp_raw, INH, T, INH, H_DIM);
  wts_kernel<<<T, 256, 0, stream>>>(iq, wp_raw, idx_wp_b, wts);
  // 13. indexer scores
  {
    dim3 g(T / 32, T / 128);
    idx_score_kernel<<<g, 256, 0, stream>>>(iq, ik, wts, sbuf);
  }
  // 14. top-k mask
  topk_kernel<<<T, 256, 0, stream>>>(sbuf, allowed);
  // 15/16. q = q_cn @ W_q_b, rope(q_pe)
  gemm(stream, q_cn, QL, W_q_b, NH * DQ, qb, NH * DQ, T, NH * DQ, QL);
  {
    int total = T * NH * (DR / 2);
    rope_kernel<<<(total + 255) / 256, 256, 0, stream>>>(
        qb, positions, NH, NH * DQ, DQ, DN, DR, total);
  }
  // 17. kv = kv_cn @ W_kv_b
  gemm(stream, kv_cn, KLAT, W_kv_b, NH * 256, kvb, NH * 256, T, NH * 256, KLAT);
  // 18. attention
  {
    dim3 g(NH, T / 32);
    attn_kernel<<<g, 256, 0, stream>>>(qb, kvb, qkv, allowed, obuf);
  }
  // 19. x1 = x + o @ W_o
  gemm(stream, obuf, NH * DV, W_o, H_DIM, x1, H_DIM, T, H_DIM, NH * DV, X);
  // 20. h2 = rmsnorm(x1, post_ln_w)
  rmsnorm_kernel<<<T, 256, 0, stream>>>(x1, H_DIM, post_ln_w, h2, H_DIM, H_DIM);
  // 21. gating
  gate_kernel<<<T, 256, 0, stream>>>(h2, W_gate, rw);
  // 22. R = 0
  zero_kernel<<<(T * H_DIM + 255) / 256, 256, 0, stream>>>(Racc, T * H_DIM);
  // 23. shared expert: R += silu(h2@Ws_gate)*(h2@Ws_up) @ Ws_down
  gemm(stream, h2, H_DIM, Ws_gate, SI_D, gbuf, SI_D, T, SI_D, H_DIM);
  gemm(stream, h2, H_DIM, Ws_up, SI_D, ubuf, SI_D, T, SI_D, H_DIM);
  silu_mul_kernel<<<(T * SI_D + 255) / 256, 256, 0, stream>>>(gbuf, ubuf, tbuf, T * SI_D);
  gemm(stream, tbuf, SI_D, Ws_down, H_DIM, Racc, H_DIM, T, H_DIM, SI_D,
       nullptr, nullptr, 0, /*accumulate=*/1);
  // 24. routed experts: R += rw[:,e] * (silu(h2@Wg_e)*(h2@Wu_e) @ Wd_e)
  for (int e = 0; e < NE; ++e) {
    const float* Wg = We_gate + (size_t)e * H_DIM * MI_D;
    const float* Wu = We_up   + (size_t)e * H_DIM * MI_D;
    const float* Wd = We_down + (size_t)e * MI_D * H_DIM;
    gemm(stream, h2, H_DIM, Wg, MI_D, gbuf, MI_D, T, MI_D, H_DIM);
    gemm(stream, h2, H_DIM, Wu, MI_D, ubuf, MI_D, T, MI_D, H_DIM);
    silu_mul_kernel<<<(T * MI_D + 255) / 256, 256, 0, stream>>>(gbuf, ubuf, tbuf, T * MI_D);
    gemm(stream, tbuf, MI_D, Wd, H_DIM, Racc, H_DIM, T, H_DIM, MI_D,
         nullptr, rw + e, NE, /*accumulate=*/1);
  }
  // 25. out = x1 + R
  add2_kernel<<<(T * H_DIM + 255) / 256, 256, 0, stream>>>(
      x1, Racc, (float*)d_out, T * H_DIM);
}
